// NRI_17643725652584
// MI455X (gfx1250) — matrix operand order repaired and checked by reading, not compiled, not run on hardware
//
#include <hip/hip_runtime.h>
#include <hip/hip_bf16.h>

// ---------------------------------------------------------------------------
// NRI forward for MI455X (gfx1250, wave32, WMMA).
//
// Mathematical reduction: enc_out_w is (H,1) => softmax over the size-1 last
// axis is identically 1.0. Hence ew==1, prob==1, so:
//   * the entire encoder is dead code w.r.t. all three outputs
//   * kl   = -63*ln(3)                    (constant)
//   * A    = ones off-diagonal            (constant)
// Only the T=8-step decoder rollout remains (~36 GFLOP, weights ~300KB ->
// L2-resident; compute/latency bound, so we feed the f16 WMMA pipe with f32
// accumulation instead of scalar VALU).
//
// Edge structure from np.where(ones-eye): edges sorted by recv; for recv=rr
// the 63 senders are j<rr ? j : j+1.  One workgroup per (b, recv) node =>
// node aggregation done in LDS, no global atomics in the hot path.
// ---------------------------------------------------------------------------

#define NB 8      // batch
#define NC 64     // nodes
#define NT_ 8     // timesteps
#define ND 10     // feature dim
#define NH 256    // hidden

typedef __attribute__((ext_vector_type(16))) _Float16 v16h;
typedef __attribute__((ext_vector_type(8)))  _Float16 v8h;
typedef __attribute__((ext_vector_type(8)))  float    v8f;

// ---- WMMA fragment helpers (wave32, 16x16x32 f16 -> f32) -------------------
// A (16x32, MxK) layout per ISA 7.12.2: lane m=lane&15, half=lane>>4,
//   element i -> k = (i&7) + 16*(i>>3) + 8*half   (two contiguous 16B runs)
__device__ __forceinline__ v16h load_fragA(const _Float16* base, int strideK,
                                           int m0, int k0) {
  const int lane = threadIdx.x & 31;
  const int m  = m0 + (lane & 15);
  const int hh = lane >> 4;
  const _Float16* p = base + m * strideK + k0 + 8 * hh;
  v8h lo = *(const v8h*)p;         // k = k0+8*hh   .. +7
  v8h hi = *(const v8h*)(p + 16);  // k = k0+16+8*hh.. +7
  v16h r;
#pragma unroll
  for (int i = 0; i < 8; ++i) { r[i] = lo[i]; r[i + 8] = hi[i]; }
  return r;
}

// B (32x16, KxN): n = lane&15, element i -> k = (i&7) + 16*(i>>3) + 8*(lane>>4).
// Weights are pre-packed so each lane's 16 halfs are contiguous (32B load).
__device__ __forceinline__ v16h load_fragB(const _Float16* packed, int frag) {
  const int lane = threadIdx.x & 31;
  return *(const v16h*)(packed + (frag * 32 + lane) * 16);
}

__device__ __forceinline__ v8f wmma_f16(v16h a, v16h b, v8f c) {
  // D = A*B + C ; emits v_wmma_f32_16x16x32_f16
  return __builtin_amdgcn_wmma_f32_16x16x32_f16(false, a, false, b,
                                                (short)0, c, false, false);
}

// ---- weight repack: f32 row-major (srcK x srcN) -> f16 B-fragment layout ---
// Padded K axis: k<kSplit -> row k; kSplit<=k<kSplit+kGap -> 0; else k-kGap.
__global__ void prep_kernel(const float* __restrict__ src,
                            _Float16* __restrict__ dst,
                            int srcK, int srcN, int KT, int NT,
                            int kSplit, int kGap) {
  int tid = blockIdx.x * 256 + threadIdx.x;
  int total = KT * NT * 512;
  if (tid >= total) return;
  int frag   = tid >> 9;
  int within = tid & 511;
  int lane = within >> 4;
  int i    = within & 15;
  int nt = frag / KT, kt = frag % KT;
  int n = nt * 16 + (lane & 15);
  int k = kt * 32 + (i & 7) + 16 * (i >> 3) + 8 * (lane >> 4);
  float v = 0.0f;
  int row = -1;
  if (k < kSplit) row = k;
  else if (k >= kSplit + kGap) row = k - kGap;
  if (row >= 0 && row < srcK && n < srcN) v = src[row * srcN + n];
  dst[tid] = (_Float16)v;   // dst index == frag*512 + lane*16 + i == tid
}

// ---- state init: state[b][c][:] = x[b][c][0][:] ----------------------------
__global__ void init_kernel(const float* __restrict__ x,
                            float* __restrict__ state) {
  int i = blockIdx.x * 256 + threadIdx.x;            // 512*10 elements
  if (i < NB * NC * ND) {
    int g = i / ND, d = i % ND;
    state[i] = x[(g * NT_ + 0) * ND + d];
  }
}

// ---- msg MLP + per-node aggregation ---------------------------------------
// One WG per (b, recv): 63 edges padded to 64 rows (row 63 masked out).
__global__ __launch_bounds__(256) void msg_kernel(
    const float* __restrict__ state, float* __restrict__ agg,
    const _Float16* __restrict__ W1p, const float* __restrict__ b1,
    const _Float16* __restrict__ W2p, const float* __restrict__ b2) {
  __shared__ __align__(16) _Float16 pre[64 * 32];    // padded K=32 (20 real)
  __shared__ __align__(16) _Float16 hbuf[64 * 256];  // layer-1 output
  __shared__ float s_agg[256];

  const int wg = blockIdx.x;            // b*64 + rr
  const int b  = wg >> 6;
  const int rr = wg & 63;
  const int tid  = threadIdx.x;
  const int lane = tid & 31;
  const int wave = tid >> 5;

  s_agg[tid] = 0.0f;

  // build pre-activations: row j = edge (send -> rr), send = j<rr ? j : j+1
  {
    int base = tid * 8;                 // 256 threads * 8 = 2048 = 64*32
    int row = base >> 5;
    int k0  = base & 31;
    const float* ss = state + (b * NC + (row < rr ? row : row + 1)) * ND;
    const float* sr = state + (b * NC + rr) * ND;
#pragma unroll
    for (int j = 0; j < 8; ++j) {
      int k = k0 + j;
      float v = 0.0f;
      if (row < 63) {
        if (k < ND)          v = ss[k];
        else if (k < 2 * ND) v = sr[k - ND];
      }
      pre[row * 32 + k] = (_Float16)v;
    }
  }
  __syncthreads();

  // layer 1: (64 x 32) @ (32 x 256), wave w owns n-tiles {2w, 2w+1}
  const int hh = lane >> 4;
#pragma unroll
  for (int s = 0; s < 2; ++s) {
    int nt = wave * 2 + s;
    int n  = nt * 16 + (lane & 15);
    float bias = b1[n];
#pragma unroll
    for (int mt = 0; mt < 4; ++mt) {
      v8f c;
#pragma unroll
      for (int j = 0; j < 8; ++j) c[j] = bias;
      v16h A = load_fragA(pre, 32, mt * 16, 0);
      v16h B = load_fragB(W1p, nt);                  // KT=1
      c = wmma_f16(A, B, c);
#pragma unroll
      for (int j = 0; j < 8; ++j) {
        float v = c[j]; v = v > 0.0f ? v : 0.0f;     // relu
        hbuf[(mt * 16 + j + 8 * hh) * 256 + n] = (_Float16)v;
      }
    }
  }
  __syncthreads();

  // layer 2: (64 x 256) @ (256 x 256); relu; masked row-sum into s_agg.
  // (reference multiplies by ew which is provably 1.0 -> omitted)
#pragma unroll
  for (int s = 0; s < 2; ++s) {
    int nt = wave * 2 + s;
    int n  = nt * 16 + (lane & 15);
    float bias = b2[n];
    v8f acc[4];
#pragma unroll
    for (int mt = 0; mt < 4; ++mt)
#pragma unroll
      for (int j = 0; j < 8; ++j) acc[mt][j] = bias;
    for (int kt = 0; kt < 8; ++kt) {
      v16h B = load_fragB(W2p, nt * 8 + kt);
#pragma unroll
      for (int mt = 0; mt < 4; ++mt) {
        v16h A = load_fragA(hbuf, 256, mt * 16, kt * 32);
        acc[mt] = wmma_f16(A, B, acc[mt]);
      }
    }
    float partial = 0.0f;
#pragma unroll
    for (int mt = 0; mt < 4; ++mt)
#pragma unroll
      for (int j = 0; j < 8; ++j) {
        int row = mt * 16 + j + 8 * hh;
        float v = acc[mt][j]; v = v > 0.0f ? v : 0.0f;
        if (row < 63) partial += v;                  // mask pad row
      }
    atomicAdd(&s_agg[n], partial);  // exactly 2 adds/addr -> commutative, det.
  }
  __syncthreads();
  agg[wg * 256 + tid] = s_agg[tid];
}

// ---- node out-MLP + residual + recon/out0 ---------------------------------
// 32 WGs, 16 nodes each.  aug = [state(10) | pad(22) | agg(256)] -> K=288.
__global__ __launch_bounds__(256) void out_kernel(
    float* __restrict__ state, const float* __restrict__ agg,
    const _Float16* __restrict__ Wo1p, const float* __restrict__ bo1,
    const _Float16* __restrict__ Wo2p, const float* __restrict__ bo2,
    const _Float16* __restrict__ Wo3p, const float* __restrict__ bo3,
    const float* __restrict__ x, int t,
    float* __restrict__ out0, float* __restrict__ recon_parts) {
  __shared__ __align__(16) _Float16 aug[16 * 288];
  __shared__ __align__(16) _Float16 h1[16 * 256];
  __shared__ __align__(16) _Float16 h2[16 * 256];
  __shared__ float s_part[32];

  const int g0   = blockIdx.x * 16;
  const int tid  = threadIdx.x;
  const int lane = tid & 31;
  const int wave = tid >> 5;
  const int hh   = lane >> 4;

  for (int e = tid; e < 16 * 288; e += 256) {
    int row = e / 288, k = e % 288;
    int g = g0 + row;
    float v = 0.0f;
    if (k < ND)       v = state[g * ND + k];
    else if (k >= 32) v = agg[g * 256 + (k - 32)];
    aug[row * 288 + k] = (_Float16)v;
  }
  __syncthreads();

  // layer o1: K=288 (9 tiles)
#pragma unroll
  for (int s = 0; s < 2; ++s) {
    int nt = wave * 2 + s;
    int n  = nt * 16 + (lane & 15);
    float bias = bo1[n];
    v8f c;
#pragma unroll
    for (int j = 0; j < 8; ++j) c[j] = bias;
    for (int kt = 0; kt < 9; ++kt) {
      v16h B = load_fragB(Wo1p, nt * 9 + kt);
      v16h A = load_fragA(aug, 288, 0, kt * 32);
      c = wmma_f16(A, B, c);
    }
#pragma unroll
    for (int j = 0; j < 8; ++j) {
      float v = c[j]; v = v > 0.0f ? v : 0.0f;
      h1[(j + 8 * hh) * 256 + n] = (_Float16)v;
    }
  }
  __syncthreads();

  // layer o2: K=256 (8 tiles)
#pragma unroll
  for (int s = 0; s < 2; ++s) {
    int nt = wave * 2 + s;
    int n  = nt * 16 + (lane & 15);
    float bias = bo2[n];
    v8f c;
#pragma unroll
    for (int j = 0; j < 8; ++j) c[j] = bias;
    for (int kt = 0; kt < 8; ++kt) {
      v16h B = load_fragB(Wo2p, nt * 8 + kt);
      v16h A = load_fragA(h1, 256, 0, kt * 32);
      c = wmma_f16(A, B, c);
    }
#pragma unroll
    for (int j = 0; j < 8; ++j) {
      float v = c[j]; v = v > 0.0f ? v : 0.0f;
      h2[(j + 8 * hh) * 256 + n] = (_Float16)v;
    }
  }
  __syncthreads();

  // layer o3: (16 x 256) @ (256 x 16[10 valid]) on wave 0 (uniform branch,
  // EXEC all-1s inside -> WMMA legal).  Residual update + recon/out0.
  if (wave == 0) {
    int n = lane & 15;
    float b3 = (n < ND) ? bo3[n] : 0.0f;
    v8f c;
#pragma unroll
    for (int j = 0; j < 8; ++j) c[j] = b3;
    for (int kt = 0; kt < 8; ++kt) {
      v16h B = load_fragB(Wo3p, kt);
      v16h A = load_fragA(h2, 256, 0, kt * 32);
      c = wmma_f16(A, B, c);
    }
    float partial = 0.0f;
    if (n < ND) {
#pragma unroll
      for (int j = 0; j < 8; ++j) {
        int g = g0 + j + 8 * hh;
        float ns = state[g * ND + n] + c[j];
        state[g * ND + n] = ns;
        if (t < NT_ - 1) {
          float dv = x[(g * NT_ + t + 1) * ND + n] - ns;
          partial += dv * dv;
        } else {
          out0[g * ND + n] = ns;   // preds[T-1] == out[:,:,-1:,:]
        }
      }
    }
    s_part[lane] = partial;
  }
  __syncthreads();
  if (tid == 0 && t < NT_ - 1) {   // deterministic fixed-order reduce
    float ssum = 0.0f;
    for (int l = 0; l < 32; ++l) ssum += s_part[l];
    recon_parts[t * 32 + blockIdx.x] = ssum;
  }
}

// ---- loss + adjacency ------------------------------------------------------
__global__ void finalize_kernel(const float* __restrict__ recon_parts,
                                float* __restrict__ outLoss,
                                float* __restrict__ outA) {
  int i = blockIdx.x * 256 + threadIdx.x;
  if (i == 0) {
    float s = 0.0f;
    for (int k = 0; k < (NT_ - 1) * 32; ++k) s += recon_parts[k];
    // kl = -63*ln(3); recon = mean over 8*64*7*10 = 35840 elements
    outLoss[0] = -69.212574186090906f + s * (1.0f / 35840.0f);
  }
  if (i < NB * NC * NC) {
    int rs = i & (NC * NC - 1);
    int r = rs >> 6, sN = rs & 63;
    outA[i] = (r == sN) ? 0.0f : 1.0f;  // edges[...,-1] == 1 (singleton softmax)
  }
}

// ---- workspace layout (bytes) ----------------------------------------------
#define OFF_RECON  0u          // 224 floats (written every launch before read)
#define OFF_STATE  1024u       // 512*10 f32   = 20480
#define OFF_AGG    21504u      // 512*256 f32  = 524288
#define OFF_W1P    545792u     // 32*256  f16  = 16384
#define OFF_W2P    562176u     // 256*256 f16  = 131072
#define OFF_WO1P   693248u     // 288*256 f16  = 147456
#define OFF_WO2P   840704u     // 256*256 f16  = 131072
#define OFF_WO3P   971776u     // 256*16  f16  = 8192   -> end 979968 (<1MB)

extern "C" void kernel_launch(void* const* d_in, const int* in_sizes, int n_in,
                              void* d_out, int out_size, void* d_ws,
                              size_t ws_size, hipStream_t stream) {
  (void)out_size; (void)ws_size;
  const float* x = (const float*)d_in[0];

  // Locate decoder params by unique sizes (dec blocks are consecutive leaves).
  int i_msg = -1, i_out = -1;
  for (int i = 0; i < n_in; ++i) {
    if (i_msg < 0 && in_sizes[i] == 2 * ND * NH)        i_msg = i;  // 20*256
    if (i_out < 0 && in_sizes[i] == (ND + NH) * NH)     i_out = i;  // 266*256
  }
  if (i_msg < 0) i_msg = 30;
  if (i_out < 0) i_out = 34;
  const float* mw1 = (const float*)d_in[i_msg + 0];
  const float* mb1 = (const float*)d_in[i_msg + 1];
  const float* mw2 = (const float*)d_in[i_msg + 2];
  const float* mb2 = (const float*)d_in[i_msg + 3];
  const float* ow1 = (const float*)d_in[i_out + 0];
  const float* ob1 = (const float*)d_in[i_out + 1];
  const float* ow2 = (const float*)d_in[i_out + 2];
  const float* ob2 = (const float*)d_in[i_out + 3];
  const float* ow3 = (const float*)d_in[i_out + 4];
  const float* ob3 = (const float*)d_in[i_out + 5];

  char* ws = (char*)d_ws;
  float*    recon = (float*)(ws + OFF_RECON);
  float*    state = (float*)(ws + OFF_STATE);
  float*    agg   = (float*)(ws + OFF_AGG);
  _Float16* W1p   = (_Float16*)(ws + OFF_W1P);
  _Float16* W2p   = (_Float16*)(ws + OFF_W2P);
  _Float16* Wo1p  = (_Float16*)(ws + OFF_WO1P);
  _Float16* Wo2p  = (_Float16*)(ws + OFF_WO2P);
  _Float16* Wo3p  = (_Float16*)(ws + OFF_WO3P);

  float* out0    = (float*)d_out;     // (B,C,1,D) = 5120
  float* outLoss = out0 + 5120;       // scalar
  float* outA    = out0 + 5121;       // (B,C,C)   = 32768

  // Repack decoder weights to f16 WMMA-B fragments (K-padding folded in).
  prep_kernel<<<32, 256, 0, stream>>>(mw1, W1p, 20, 256, 1, 16, 20, 12);
  prep_kernel<<<256, 256, 0, stream>>>(mw2, W2p, 256, 256, 8, 16, 256, 0);
  prep_kernel<<<288, 256, 0, stream>>>(ow1, Wo1p, 266, 256, 9, 16, 10, 22);
  prep_kernel<<<256, 256, 0, stream>>>(ow2, Wo2p, 256, 256, 8, 16, 256, 0);
  prep_kernel<<<16, 256, 0, stream>>>(ow3, Wo3p, 256, 10, 8, 1, 256, 0);

  init_kernel<<<20, 256, 0, stream>>>(x, state);

  for (int t = 0; t < NT_; ++t) {
    msg_kernel<<<NB * NC, 256, 0, stream>>>(state, agg, W1p, mb1, W2p, mb2);
    out_kernel<<<NB * NC / 16, 256, 0, stream>>>(state, agg, Wo1p, ob1,
                                                 Wo2p, ob2, Wo3p, ob3,
                                                 x, t, out0, recon);
  }
  finalize_kernel<<<128, 256, 0, stream>>>(recon, outLoss, outA);
}
